// TextEncoderSepPointEnrich_42623255446297
// MI455X (gfx1250) — matrix operand order repaired and checked by reading, not compiled, not run on hardware
//
#include <hip/hip_runtime.h>
#include <math.h>

// ---------------- CDNA5 WMMA types ----------------
typedef __attribute__((ext_vector_type(16))) __bf16 v16bf;
typedef __attribute__((ext_vector_type(8)))  float  v8f;

#define B_   32
#define L_   128
#define NPT  1024
#define E_   768
#define H_   512
#define P_   256
#define NH_  4
#define HD_  128
#define G4H  2048
#define SCALE_ 0.08838834764831845f   // 1/sqrt(128)

union Frag16 { v16bf v; uint4 q[2]; };

__device__ __forceinline__ unsigned short f2bf(float f) {
  // round-to-nearest-even fp32 -> bf16 (bit manipulation: no __bf16 arithmetic needed)
  unsigned int u = __float_as_uint(f);
  unsigned int r = (u + 0x7fffu + ((u >> 16) & 1u)) >> 16;
  return (unsigned short)r;
}

__device__ __forceinline__ float sigf(float x) { return 1.0f / (1.0f + __expf(-x)); }
// branchless tanh (no exec-mask divergence on the LSTM serial path):
// tanh(x) = 1 - 2/(exp(2x)+1);  exp underflow/overflow saturate to -1/+1 correctly.
__device__ __forceinline__ float tanh_fast(float x) { return 1.0f - 2.0f / (__expf(2.0f * x) + 1.0f); }

__device__ __forceinline__ v8f wmma_bf16(const Frag16& a, const Frag16& b, v8f c) {
  return __builtin_amdgcn_wmma_f32_16x16x32_bf16(false, a.v, false, b.v, (short)0, c, false, false);
}

// ---------------- weight transpose + bf16 convert:  W(K,N) fp32 -> Wt(N, out_ld) bf16 ----------------
__global__ void transpose_bf16_kernel(const float* __restrict__ W, unsigned short* __restrict__ Wt,
                                      int K, int N, int out_ld, int col_off,
                                      long inStride, long outStride) {
  __shared__ float tile[32][33];
  int z = blockIdx.z;
  const float* Wz = W + (long)z * inStride;
  unsigned short* Oz = Wt + (long)z * outStride;
  int n0 = blockIdx.x * 32, k0 = blockIdx.y * 32;
  int tx = threadIdx.x, ty = threadIdx.y;
  #pragma unroll
  for (int i = 0; i < 4; i++)
    tile[ty + 8 * i][tx] = Wz[(long)(k0 + ty + 8 * i) * N + n0 + tx];
  __syncthreads();
  #pragma unroll
  for (int i = 0; i < 4; i++)
    Oz[(long)(n0 + ty + 8 * i) * out_ld + col_off + k0 + tx] = f2bf(tile[tx][ty + 8 * i]);
}

__global__ void add_bias_vec(const float* __restrict__ a, const float* __restrict__ b,
                             float* __restrict__ o, int n) {
  int i = blockIdx.x * 256 + threadIdx.x;
  if (i < n) o[i] = a[i] + b[i];
}

// ---------------- generic WMMA GEMM:  C(M,N) = act(A(M,K) @ Bt(N,K)^T + bias) ----------------
// 256 threads = 8 waves; BM=128 BN=128 BK=32; wave w owns M-tile w x all 8 N-tiles.
template<int ACT>  // 0 none, 1 leaky_relu(0.01), 2 relu
__launch_bounds__(256)
__global__ void gemm_bt(const float* __restrict__ A, long sA,
                        const unsigned short* __restrict__ Bt, long sB,
                        const float* __restrict__ bias, long sBias,
                        float* __restrict__ C, long sC,
                        int M, int N, int K) {
  __shared__ unsigned short As[128 * 32];
  __shared__ unsigned short Bs[128 * 32];
  int z = blockIdx.z;
  A += (long)z * sA; Bt += (long)z * sB; bias += (long)z * sBias; C += (long)z * sC;
  int m0 = blockIdx.x * 128, n0 = blockIdx.y * 128;
  int tid = threadIdx.x, lane = tid & 31, w = tid >> 5, ln = lane & 15, half = lane >> 4;

  v8f acc[8];
  v8f vzero = {0.f, 0.f, 0.f, 0.f, 0.f, 0.f, 0.f, 0.f};
  #pragma unroll
  for (int i = 0; i < 8; i++) acc[i] = vzero;

  for (int k0 = 0; k0 < K; k0 += 32) {
    // stage A tile (fp32 -> bf16), zero-fill rows >= M
    #pragma unroll
    for (int jj = 0; jj < 4; jj++) {
      int c = jj * 256 + tid;
      int row = c >> 3, colc = (c & 7) * 4;
      float4 v = make_float4(0.f, 0.f, 0.f, 0.f);
      if (m0 + row < M) v = *(const float4*)&A[(long)(m0 + row) * K + k0 + colc];
      unsigned int p0 = (unsigned int)f2bf(v.x) | ((unsigned int)f2bf(v.y) << 16);
      unsigned int p1 = (unsigned int)f2bf(v.z) | ((unsigned int)f2bf(v.w) << 16);
      *(uint2*)&As[row * 32 + colc] = make_uint2(p0, p1);
    }
    // stage B tile (already bf16)
    #pragma unroll
    for (int jj = 0; jj < 2; jj++) {
      int c = jj * 256 + tid;
      int row = c >> 2, colc = (c & 3) * 8;
      *(uint4*)&Bs[row * 32 + colc] = *(const uint4*)&Bt[(long)(n0 + row) * K + k0 + colc];
    }
    if (k0 + 32 < K)  // prefetch next B K-slice (global_prefetch_b8)
      __builtin_prefetch(&Bt[(long)(n0 + (tid >> 1)) * K + k0 + 32], 0, 0);
    __syncthreads();

    Frag16 a;
    int m = w * 16 + ln;
    a.q[0] = *(const uint4*)&As[m * 32 + half * 8];
    a.q[1] = *(const uint4*)&As[m * 32 + 16 + half * 8];
    #pragma unroll
    for (int nt = 0; nt < 8; nt++) {
      Frag16 b;
      int n = nt * 16 + ln;
      b.q[0] = *(const uint4*)&Bs[n * 32 + half * 8];
      b.q[1] = *(const uint4*)&Bs[n * 32 + 16 + half * 8];
      acc[nt] = wmma_bf16(a, b, acc[nt]);
    }
    __syncthreads();
  }

  #pragma unroll
  for (int nt = 0; nt < 8; nt++) {
    int col = n0 + nt * 16 + ln;
    float bb = bias[col];
    #pragma unroll
    for (int r = 0; r < 8; r++) {
      int row = m0 + w * 16 + half * 8 + r;
      if (row < M) {
        float v = acc[nt][r] + bb;
        if (ACT == 1) v = v > 0.f ? v : 0.01f * v;
        if (ACT == 2) v = v > 0.f ? v : 0.f;
        C[(long)row * N + col] = v;
      }
    }
  }
}

// ---------------- LSTM layer: persistent per-text-block recurrence ----------------
// gates(32x2048) = [xt | h](32x1024) @ Wcomb^T(2048x1024).  h kept bf16 in LDS, c in regs.
// 1024 thr = 32 waves; wave w owns columns j=w*16+ln of H for ALL four gates, so the
// nonlinearity is register-local. Gates are processed SEQUENTIALLY (g -> i -> f -> o)
// so only one accumulator pair is live at a time (no scratch spills at 8 waves/SIMD):
//   tmp = tanh(g); tmp = sig(i)*tmp; c = sig(f)*c + tmp; h = sig(o)*tanh(c)
#define GATE_KLOOP(G, ACC0, ACC1)                                              \
  {                                                                            \
    const unsigned short* Wg = W + (long)((G) * H_ + j) * 1024 + half * 8;     \
    _Pragma("unroll 4")                                                        \
    for (int ks = 0; ks < 32; ks++) {                                          \
      Frag16 a0, a1, b;                                                        \
      int base0 = ln * 1024 + ks * 32;                                         \
      a0.q[0] = *(const uint4*)&Abuf[base0 + half * 8];                        \
      a0.q[1] = *(const uint4*)&Abuf[base0 + 16 + half * 8];                   \
      a1.q[0] = *(const uint4*)&Abuf[base0 + 16384 + half * 8];                \
      a1.q[1] = *(const uint4*)&Abuf[base0 + 16384 + 16 + half * 8];           \
      b.q[0] = *(const uint4*)(Wg + ks * 32);                                  \
      b.q[1] = *(const uint4*)(Wg + ks * 32 + 16);                             \
      ACC0 = wmma_bf16(a0, b, ACC0);                                           \
      ACC1 = wmma_bf16(a1, b, ACC1);                                           \
    }                                                                          \
  }

__launch_bounds__(1024)
__global__ void lstm_layer(const float* __restrict__ xseq,        // (5,32,128,512) fp32
                           const unsigned short* __restrict__ wt, // (5,2,2048,1024) bf16
                           const float* __restrict__ bsum,        // (5,2,2048) fp32
                           float* __restrict__ hseq,              // (5,32,128,512) fp32
                           int layer) {
  __shared__ unsigned short Abuf[32 * 1024];   // 64KB: cols [0,512)=xt, [512,1024)=h
  int tb = blockIdx.x;
  const unsigned short* W = wt + (long)(tb * 2 + layer) * G4H * 1024;
  const float* bs = bsum + (long)(tb * 2 + layer) * G4H;
  const float* xs = xseq + (long)tb * B_ * L_ * H_;
  float* hs = hseq + (long)tb * B_ * L_ * H_;
  int tid = threadIdx.x, lane = tid & 31, w = tid >> 5, ln = lane & 15, half = lane >> 4;
  int j = w * 16 + ln;   // column within H owned by this lane

  float bg_i = bs[0 * H_ + j];
  float bg_f = bs[1 * H_ + j];
  float bg_g = bs[2 * H_ + j];
  float bg_o = bs[3 * H_ + j];

  float cacc[2][8];
  float tmp[2][8];
  #pragma unroll
  for (int mt = 0; mt < 2; mt++)
    #pragma unroll
    for (int r = 0; r < 8; r++) cacc[mt][r] = 0.f;

  for (int i = tid; i < 32 * 1024; i += 1024) Abuf[i] = 0;   // h0 = 0
  __syncthreads();

  v8f vzero = {0.f, 0.f, 0.f, 0.f, 0.f, 0.f, 0.f, 0.f};

  for (int t = 0; t < L_; t++) {
    // stage xt (fp32 -> bf16) into cols [0,512): 4 x float4 loads + 4 x ds_store_b64 per thread
    #pragma unroll
    for (int jj = 0; jj < 4; jj++) {
      int c = jj * 1024 + tid;
      int row = c >> 7, colc = (c & 127) * 4;
      float4 v = *(const float4*)&xs[((long)row * L_ + t) * H_ + colc];
      unsigned int p0 = (unsigned int)f2bf(v.x) | ((unsigned int)f2bf(v.y) << 16);
      unsigned int p1 = (unsigned int)f2bf(v.z) | ((unsigned int)f2bf(v.w) << 16);
      *(uint2*)&Abuf[row * 1024 + colc] = make_uint2(p0, p1);
    }
    __syncthreads();

    v8f acc0, acc1;

    // gate g (cell input): tmp = tanh(.)
    acc0 = vzero; acc1 = vzero;
    GATE_KLOOP(2, acc0, acc1);
    #pragma unroll
    for (int r = 0; r < 8; r++) { tmp[0][r] = tanh_fast(acc0[r] + bg_g); tmp[1][r] = tanh_fast(acc1[r] + bg_g); }

    // gate i: tmp = sig(i)*tmp
    acc0 = vzero; acc1 = vzero;
    GATE_KLOOP(0, acc0, acc1);
    #pragma unroll
    for (int r = 0; r < 8; r++) { tmp[0][r] *= sigf(acc0[r] + bg_i); tmp[1][r] *= sigf(acc1[r] + bg_i); }

    // gate f: c = sig(f)*c + tmp
    acc0 = vzero; acc1 = vzero;
    GATE_KLOOP(1, acc0, acc1);
    #pragma unroll
    for (int r = 0; r < 8; r++) {
      cacc[0][r] = sigf(acc0[r] + bg_f) * cacc[0][r] + tmp[0][r];
      cacc[1][r] = sigf(acc1[r] + bg_f) * cacc[1][r] + tmp[1][r];
    }

    // gate o: h = sig(o)*tanh(c)  (accumulate before barrier, apply after)
    acc0 = vzero; acc1 = vzero;
    GATE_KLOOP(3, acc0, acc1);
    __syncthreads();   // all waves done reading Abuf this step

    #pragma unroll
    for (int r = 0; r < 8; r++) {
      float h0 = sigf(acc0[r] + bg_o) * tanh_fast(cacc[0][r]);
      float h1 = sigf(acc1[r] + bg_o) * tanh_fast(cacc[1][r]);
      int m0r = half * 8 + r;          // batch rows for mt=0 / mt=1
      int m1r = 16 + half * 8 + r;
      Abuf[m0r * 1024 + H_ + j] = f2bf(h0);
      Abuf[m1r * 1024 + H_ + j] = f2bf(h1);
      hs[((long)m0r * L_ + t) * H_ + j] = h0;
      hs[((long)m1r * L_ + t) * H_ + j] = h1;
    }
    __syncthreads();   // h visible before next step
  }
}

// ---------------- gather last valid timestep ----------------
__global__ void gather_last(const float* __restrict__ h1seq, const int* __restrict__ text_length,
                            float* __restrict__ lastbuf) {
  int tb = blockIdx.x, b = blockIdx.y, tid = threadIdx.x;   // 512 threads
  int len = text_length[b * 5 + tb];
  len = len < 1 ? 1 : (len > L_ ? L_ : len);
  lastbuf[((long)tb * B_ + b) * H_ + tid] =
      h1seq[(((long)tb * B_ + b) * L_ + (len - 1)) * H_ + tid];
}

// ---------------- fused factored attention: scores + softmax + weighted-point context ----------------
// scores[b,h,n] = (Wk_head^T q) . pts[b,n,:] * SCALE  (q.bk shift is softmax-invariant)
// ctx[s,h,b,:]  = sum_n attn[n] * pts[b,n,:]
__launch_bounds__(256)
__global__ void seg_attention(const float* __restrict__ qbuf,  // (4,32,512)
                              const float* __restrict__ Wk,    // (4,256,512)
                              const float* __restrict__ pts,   // (32,1024,256)
                              const int* __restrict__ mask,    // (32,1024)
                              float* __restrict__ ctx,         // (4,4,32,256)
                              int* __restrict__ anyvalid) {    // (4,32)
  int b = blockIdx.x, h = blockIdx.y, s = blockIdx.z;
  __shared__ float qs[HD_];
  __shared__ float qk[P_];
  __shared__ float sc[NPT];
  __shared__ float red[256];
  int tid = threadIdx.x;
  if (tid < HD_) qs[tid] = qbuf[((long)s * B_ + b) * H_ + h * HD_ + tid];
  __syncthreads();
  {
    const float* wr = Wk + ((long)s * P_ + tid) * H_ + h * HD_;
    float a = 0.f;
    for (int d = 0; d < HD_; d++) a += qs[d] * wr[d];
    qk[tid] = a;
  }
  __syncthreads();
  int wv = tid >> 5, lane = tid & 31;
  for (int n = wv; n < NPT; n += 8) {
    const float* pr = pts + ((long)b * NPT + n) * P_ + lane * 8;
    float part = 0.f;
    #pragma unroll
    for (int q = 0; q < 8; q++) part += pr[q] * qk[lane * 8 + q];
    #pragma unroll
    for (int off = 16; off >= 1; off >>= 1) part += __shfl_xor(part, off);
    if (lane == 0) {
      bool valid = (mask[b * NPT + n] == s + 1);
      sc[n] = valid ? part * SCALE_ : -1e9f;
    }
  }
  __syncthreads();
  float mx = -1e30f;
  for (int i = tid; i < NPT; i += 256) mx = fmaxf(mx, sc[i]);
  red[tid] = mx; __syncthreads();
  for (int st = 128; st >= 1; st >>= 1) { if (tid < st) red[tid] = fmaxf(red[tid], red[tid + st]); __syncthreads(); }
  mx = red[0]; __syncthreads();
  float sum = 0.f, cnt = 0.f;
  for (int i = tid; i < NPT; i += 256) {
    if (sc[i] > -1e8f) cnt += 1.f;
    float e = __expf(sc[i] - mx);
    sc[i] = e; sum += e;
  }
  red[tid] = sum; __syncthreads();
  for (int st = 128; st >= 1; st >>= 1) { if (tid < st) red[tid] += red[tid + st]; __syncthreads(); }
  sum = red[0]; __syncthreads();
  red[tid] = cnt; __syncthreads();
  for (int st = 128; st >= 1; st >>= 1) { if (tid < st) red[tid] += red[tid + st]; __syncthreads(); }
  cnt = red[0]; __syncthreads();
  float inv = 1.0f / sum;
  for (int i = tid; i < NPT; i += 256) sc[i] *= inv;
  __syncthreads();
  {
    float a = 0.f;
    for (int n = 0; n < NPT; n++) a += sc[n] * pts[((long)b * NPT + n) * P_ + tid];
    ctx[(((long)s * NH_ + h) * B_ + b) * P_ + tid] = a;
  }
  if (tid == 0 && h == 0) anyvalid[s * B_ + b] = (cnt > 0.5f) ? 1 : 0;
}

// ---------------- (s,h,b,d) -> (s,b,h*d) head merge ----------------
__global__ void head_merge(const float* __restrict__ headout, float* __restrict__ attn_in) {
  int i = blockIdx.x * 256 + threadIdx.x;   // 65536 total
  int s = i >> 14, rem = i & 16383, b = rem >> 9, c = rem & 511;
  int h = c >> 7, d = c & 127;
  attn_in[i] = headout[(((s * NH_ + h) * B_ + b) << 7) + d];
}

// ---------------- classifier head ----------------
__global__ void classifier(const float* __restrict__ tg,
                           const float* __restrict__ W1, const float* __restrict__ b1,
                           const float* __restrict__ W2, const float* __restrict__ b2,
                           const float* __restrict__ W3, const float* __restrict__ b3,
                           float* __restrict__ out_cls) {
  int b = blockIdx.x, tid = threadIdx.x;   // 128 threads
  __shared__ float h1[128];
  __shared__ float h2[16];
  const float* x = tg + (long)b * H_;
  float a = b1[tid];
  for (int k = 0; k < H_; k++) a += x[k] * W1[k * 128 + tid];
  h1[tid] = fmaxf(a, 0.f);
  __syncthreads();
  if (tid < 16) {
    float a2 = b2[tid];
    for (int k = 0; k < 128; k++) a2 += h1[k] * W2[k * 16 + tid];
    h2[tid] = fmaxf(a2, 0.f);
  }
  __syncthreads();
  if (tid < 3) {
    float a3 = b3[tid];
    for (int k = 0; k < 16; k++) a3 += h2[k] * W3[k * 3 + tid];
    out_cls[b * 3 + tid] = a3;
  }
}

// ---------------- output assembly ----------------
__global__ void epilogue(const float* __restrict__ blk_out, const float* __restrict__ proj,
                         const int* __restrict__ anyvalid, float* __restrict__ d_out) {
  int i = blockIdx.x * 256 + threadIdx.x;
  if (i < B_ * H_) {
    float v = blk_out[i];
    d_out[i] = v;                                // text_general
    d_out[5 * B_ * H_ + B_ * 3 + i] = v;         // cls_feat
  }
  if (i < 4 * B_ * H_) {
    int s = i >> 14, b = (i >> 9) & 31;
    float v = blk_out[B_ * H_ + i];
    if (anyvalid[s * B_ + b]) v += proj[i];
    d_out[B_ * H_ + i] = v;                      // segs
  }
}

// ==================================================================================
extern "C" void kernel_launch(void* const* d_in, const int* in_sizes, int n_in,
                              void* d_out, int out_size, void* d_ws, size_t ws_size,
                              hipStream_t stream) {
  (void)in_sizes; (void)n_in; (void)out_size; (void)ws_size;
  const float* x_txt[5] = {(const float*)d_in[0], (const float*)d_in[1], (const float*)d_in[2],
                           (const float*)d_in[3], (const float*)d_in[4]};
  const int*   text_length = (const int*)d_in[5];
  const float* radar  = (const float*)d_in[6];
  const int*   rmask  = (const int*)d_in[7];
  const float* blk_W1 = (const float*)d_in[8];
  const float* blk_b1 = (const float*)d_in[9];
  const float* blk_W2 = (const float*)d_in[10];
  const float* blk_b2 = (const float*)d_in[11];
  const float* lstm_Wih = (const float*)d_in[12];
  const float* lstm_Whh = (const float*)d_in[13];
  const float* lstm_bih = (const float*)d_in[14];
  const float* lstm_bhh = (const float*)d_in[15];
  const float* blk_lin_W = (const float*)d_in[16];
  const float* blk_lin_b = (const float*)d_in[17];
  const float* att_Wq = (const float*)d_in[18];
  const float* att_bq = (const float*)d_in[19];
  const float* att_Wk = (const float*)d_in[20];
  const float* att_Wv = (const float*)d_in[22];
  const float* att_bv = (const float*)d_in[23];
  const float* att_Wo = (const float*)d_in[24];
  const float* att_bo = (const float*)d_in[25];
  const float* cls_W1 = (const float*)d_in[26];
  const float* cls_b1 = (const float*)d_in[27];
  const float* cls_W2 = (const float*)d_in[28];
  const float* cls_b2 = (const float*)d_in[29];
  const float* cls_W3 = (const float*)d_in[30];
  const float* cls_b3 = (const float*)d_in[31];

  char* ws = (char*)d_ws;
  size_t off = 0;
  auto alloc = [&](size_t bytes) -> char* {
    char* p = ws + off;
    off += (bytes + 255) & ~(size_t)255;
    return p;
  };
  unsigned short* wt_mlp1 = (unsigned short*)alloc(5L * 512 * 768 * 2);
  unsigned short* wt_mlp2 = (unsigned short*)alloc(5L * 512 * 512 * 2);
  unsigned short* wt_lstm = (unsigned short*)alloc(10L * 2048 * 1024 * 2);
  unsigned short* wt_lin  = (unsigned short*)alloc(5L * 512 * 512 * 2);
  unsigned short* wt_q    = (unsigned short*)alloc(4L * 512 * 512 * 2);
  unsigned short* wt_v    = (unsigned short*)alloc(4L * 512 * 256 * 2);
  unsigned short* wt_o    = (unsigned short*)alloc(4L * 512 * 512 * 2);
  float* bsum    = (float*)alloc(10L * 2048 * 4);
  float* bufA    = (float*)alloc(5L * 4096 * 512 * 4);   // mlp_tmp, then h0seq
  float* bufB    = (float*)alloc(5L * 4096 * 512 * 4);   // mlpH, then h1seq
  float* lastbuf = (float*)alloc(5L * 32 * 512 * 4);
  float* blk_out = (float*)alloc(5L * 32 * 512 * 4);
  float* qbuf    = (float*)alloc(4L * 32 * 512 * 4);
  float* ctx     = (float*)alloc(4L * 4 * 32 * 256 * 4);
  float* headout = (float*)alloc(16L * 32 * 128 * 4);
  float* attn_in = (float*)alloc(4L * 32 * 512 * 4);
  float* proj    = (float*)alloc(4L * 32 * 512 * 4);
  int*   anyv    = (int*)alloc(4L * 32 * 4);

  dim3 tb32(32, 8);
  // weight transposes -> bf16 (N,K)
  transpose_bf16_kernel<<<dim3(16, 24, 5), tb32, 0, stream>>>(blk_W1, wt_mlp1, 768, 512, 768, 0, 768L * 512, 512L * 768);
  transpose_bf16_kernel<<<dim3(16, 16, 5), tb32, 0, stream>>>(blk_W2, wt_mlp2, 512, 512, 512, 0, 512L * 512, 512L * 512);
  transpose_bf16_kernel<<<dim3(64, 16, 10), tb32, 0, stream>>>(lstm_Wih, wt_lstm, 512, 2048, 1024, 0, 512L * 2048, 2048L * 1024);
  transpose_bf16_kernel<<<dim3(64, 16, 10), tb32, 0, stream>>>(lstm_Whh, wt_lstm, 512, 2048, 1024, 512, 512L * 2048, 2048L * 1024);
  transpose_bf16_kernel<<<dim3(16, 16, 5), tb32, 0, stream>>>(blk_lin_W, wt_lin, 512, 512, 512, 0, 512L * 512, 512L * 512);
  transpose_bf16_kernel<<<dim3(16, 16, 4), tb32, 0, stream>>>(att_Wq, wt_q, 512, 512, 512, 0, 512L * 512, 512L * 512);
  transpose_bf16_kernel<<<dim3(16, 8, 4), tb32, 0, stream>>>(att_Wv, wt_v, 256, 512, 256, 0, 256L * 512, 512L * 256);
  transpose_bf16_kernel<<<dim3(16, 16, 4), tb32, 0, stream>>>(att_Wo, wt_o, 512, 512, 512, 0, 512L * 512, 512L * 512);
  add_bias_vec<<<80, 256, 0, stream>>>(lstm_bih, lstm_bhh, bsum, 10 * 2048);

  // MLP: leaky(x@W1+b1) @ W2 + b2
  for (int tb = 0; tb < 5; tb++)
    gemm_bt<1><<<dim3(32, 4, 1), 256, 0, stream>>>(x_txt[tb], 0, wt_mlp1 + (long)tb * 512 * 768, 0,
                                                   blk_b1 + tb * 512, 0, bufA + (long)tb * 4096 * 512, 0,
                                                   4096, 512, 768);
  gemm_bt<0><<<dim3(32, 4, 5), 256, 0, stream>>>(bufA, 4096L * 512, wt_mlp2, 512L * 512,
                                                 blk_b2, 512, bufB, 4096L * 512, 4096, 512, 512);
  // LSTM: layer0 (bufB -> bufA), layer1 (bufA -> bufB)
  lstm_layer<<<5, 1024, 0, stream>>>(bufB, wt_lstm, bsum, bufA, 0);
  lstm_layer<<<5, 1024, 0, stream>>>(bufA, wt_lstm, bsum, bufB, 1);
  // last valid step + linear head
  gather_last<<<dim3(5, 32), 512, 0, stream>>>(bufB, text_length, lastbuf);
  gemm_bt<0><<<dim3(1, 4, 5), 256, 0, stream>>>(lastbuf, 32L * 512, wt_lin, 512L * 512,
                                                blk_lin_b, 512, blk_out, 32L * 512, 32, 512, 512);
  // attention: q projection, fused factored score/softmax/context, V and O projections
  gemm_bt<0><<<dim3(1, 4, 4), 256, 0, stream>>>(blk_out + 32 * 512, 32L * 512, wt_q, 512L * 512,
                                                att_bq, 512, qbuf, 32L * 512, 32, 512, 512);
  seg_attention<<<dim3(32, 4, 4), 256, 0, stream>>>(qbuf, att_Wk, radar, rmask, ctx, anyv);
  gemm_bt<0><<<dim3(1, 1, 16), 256, 0, stream>>>(ctx, 32L * 256, wt_v, 128L * 256,
                                                 att_bv, 128, headout, 32L * 128, 32, 128, 256);
  head_merge<<<256, 256, 0, stream>>>(headout, attn_in);
  gemm_bt<0><<<dim3(1, 4, 4), 256, 0, stream>>>(attn_in, 32L * 512, wt_o, 512L * 512,
                                                att_bo, 512, proj, 32L * 512, 32, 512, 512);
  // heads + outputs
  classifier<<<32, 128, 0, stream>>>(blk_out, cls_W1, cls_b1, cls_W2, cls_b2, cls_W3, cls_b3,
                                     (float*)d_out + 5 * B_ * H_);
  epilogue<<<256, 256, 0, stream>>>(blk_out, proj, anyv, (float*)d_out);
}
